// TransformerBlock_cross_8993661518348
// MI455X (gfx1250) — matrix operand order repaired and checked by reading, not compiled, not run on hardware
//
#include <hip/hip_runtime.h>
#include <hip/hip_bf16.h>
#include <math.h>

// Problem constants (match reference)
#define HWPIX 65536
#define WDIM 256
#define HDIM 256
#define BN 2
#define CN 64
#define NHN 8
#define HIDN 170

typedef __attribute__((ext_vector_type(16))) __bf16 v16bf;
typedef __attribute__((ext_vector_type(8)))  float  v8f;

__device__ __forceinline__ __bf16 to_bf(float x) { return (__bf16)x; }

__device__ __forceinline__ float wave_reduce_sum(float v) {
#pragma unroll
  for (int off = 16; off > 0; off >>= 1)
    v += __shfl_xor(v, off, 32);
  return v;
}

// ---------------------------------------------------------------------------
// LayerNorm over channel dim (per pixel), NCHW.
// ---------------------------------------------------------------------------
__global__ void ln_ch_kernel(const float* __restrict__ in,
                             const float* __restrict__ w,
                             const float* __restrict__ b,
                             float* __restrict__ out) {
  int bi = blockIdx.z;
  int p = blockIdx.x * blockDim.x + threadIdx.x;
  const float* src = in + (size_t)bi * CN * HWPIX + p;
  float* dst = out + (size_t)bi * CN * HWPIX + p;
  float vals[CN];
  float s = 0.f, s2 = 0.f;
#pragma unroll
  for (int c = 0; c < CN; ++c) {
    float v = src[(size_t)c * HWPIX];
    vals[c] = v;
    s += v;
    s2 += v * v;
  }
  float mu = s * (1.0f / CN);
  float var = s2 * (1.0f / CN) - mu * mu;
  float inv = rsqrtf(var + 1e-5f);
#pragma unroll
  for (int c = 0; c < CN; ++c)
    dst[(size_t)c * HWPIX] = (vals[c] - mu) * inv * w[c] + b[c];
}

// ---------------------------------------------------------------------------
// Zero-pad a weight matrix [Mreal,Kreal] -> [Mpad,Kpad] (row-major) so the
// WMMA GEMM can run completely guard-free.
// ---------------------------------------------------------------------------
__global__ void pad_weight_kernel(const float* __restrict__ in,
                                  float* __restrict__ out,
                                  int Mreal, int Kreal, int Kpad, int total) {
  int i = blockIdx.x * blockDim.x + threadIdx.x;
  if (i >= total) return;
  int r = i / Kpad, c = i % Kpad;
  out[i] = (r < Mreal && c < Kreal) ? in[r * Kreal + c] : 0.f;
}

// ---------------------------------------------------------------------------
// 1x1 conv as WMMA GEMM:  Y[b, m, n] = sum_k W[m,k] * X[b, k, n] (+ R)
// Guard-free: W is [M,KPAD] exactly (pre-padded), M % 32 == 0, KPAD % 32 == 0.
// Each wave computes TWO 16x16 tiles (m0, m0+16) sharing one B fragment.
// Fragment layouts per ISA 7.12.2:
//   A (16x32 bf16): row = lane&15; lane<16 holds K 0-7/16-23, lane>=16 K 8-15/24-31
//   B (32x16 bf16): col = lane&15; lane<16 holds K 0-7/16-23, lane>=16 K 8-15/24-31
//   C/D (f32): VGPR r -> row m0 + r + 8*(lane>>4), col = n0 + (lane&15)
// ---------------------------------------------------------------------------
template <int KPAD, bool HAS_R>
__global__ void gemm_wmma_bf16(const float* __restrict__ W,
                               const float* __restrict__ X,
                               float* __restrict__ Y,
                               const float* __restrict__ R,
                               int M) {
  int lane = threadIdx.x & 31;
  int wave = threadIdx.x >> 5;
  int n0 = (blockIdx.x * (blockDim.x >> 5) + wave) * 16;
  int m0 = blockIdx.y * 32;
  int bi = blockIdx.z;

  const float* Xb = X + (size_t)bi * KPAD * HWPIX;
  float* Yb = Y + (size_t)bi * M * HWPIX;
  const float* Rb = HAS_R ? (R + (size_t)bi * M * HWPIX) : nullptr;

  int col = n0 + (lane & 15);
  int hi = lane >> 4;  // 0 or 1
  int rowA = m0 + (lane & 15);

  v8f acc0 = {}, acc1 = {};
#pragma unroll
  for (int kk = 0; kk < KPAD; kk += 32) {
    if (kk + 32 < KPAD) {
      // pull next B slab toward cache (global_prefetch_b8)
      __builtin_prefetch(&Xb[(size_t)(kk + 32 + hi * 16) * HWPIX + col], 0, 0);
    }
    // A fragments: 8+8 consecutive floats per half -> b128 loads
    const float* w0 = W + (size_t)rowA * KPAD + kk + hi * 8;
    const float* w1 = w0 + (size_t)16 * KPAD;
    float4 p00 = *(const float4*)(w0);
    float4 p01 = *(const float4*)(w0 + 4);
    float4 p02 = *(const float4*)(w0 + 16);
    float4 p03 = *(const float4*)(w0 + 20);
    float4 p10 = *(const float4*)(w1);
    float4 p11 = *(const float4*)(w1 + 4);
    float4 p12 = *(const float4*)(w1 + 16);
    float4 p13 = *(const float4*)(w1 + 20);
    v16bf a0, a1, bm;
    a0[0] = to_bf(p00.x);  a0[1] = to_bf(p00.y);
    a0[2] = to_bf(p00.z);  a0[3] = to_bf(p00.w);
    a0[4] = to_bf(p01.x);  a0[5] = to_bf(p01.y);
    a0[6] = to_bf(p01.z);  a0[7] = to_bf(p01.w);
    a0[8] = to_bf(p02.x);  a0[9] = to_bf(p02.y);
    a0[10] = to_bf(p02.z); a0[11] = to_bf(p02.w);
    a0[12] = to_bf(p03.x); a0[13] = to_bf(p03.y);
    a0[14] = to_bf(p03.z); a0[15] = to_bf(p03.w);
    a1[0] = to_bf(p10.x);  a1[1] = to_bf(p10.y);
    a1[2] = to_bf(p10.z);  a1[3] = to_bf(p10.w);
    a1[4] = to_bf(p11.x);  a1[5] = to_bf(p11.y);
    a1[6] = to_bf(p11.z);  a1[7] = to_bf(p11.w);
    a1[8] = to_bf(p12.x);  a1[9] = to_bf(p12.y);
    a1[10] = to_bf(p12.z); a1[11] = to_bf(p12.w);
    a1[12] = to_bf(p13.x); a1[13] = to_bf(p13.y);
    a1[14] = to_bf(p13.z); a1[15] = to_bf(p13.w);
#pragma unroll
    for (int j = 0; j < 16; ++j)
      bm[j] = to_bf(Xb[(size_t)(kk + hi * 8 + j + (j >= 8 ? 8 : 0)) * HWPIX + col]);
    acc0 = __builtin_amdgcn_wmma_f32_16x16x32_bf16(false, a0, false, bm,
                                                   (short)0, acc0, false, false);
    acc1 = __builtin_amdgcn_wmma_f32_16x16x32_bf16(false, a1, false, bm,
                                                   (short)0, acc1, false, false);
  }
  size_t base = (size_t)(m0 + hi * 8) * HWPIX + col;
#pragma unroll
  for (int r = 0; r < 8; ++r) {
    size_t idx = base + (size_t)r * HWPIX;
    float v = acc0[r];
    if (HAS_R) v += Rb[idx];
    Yb[idx] = v;
    size_t idx2 = idx + (size_t)16 * HWPIX;
    float v2 = acc1[r];
    if (HAS_R) v2 += Rb[idx2];
    Yb[idx2] = v2;
  }
}

// ---------------------------------------------------------------------------
// Depthwise 3x3, SAME zero padding, cross-correlation (XLA conv semantics).
// ---------------------------------------------------------------------------
__global__ void dwconv3_kernel(const float* __restrict__ in,
                               const float* __restrict__ w,
                               float* __restrict__ out,
                               int inRows, int outRows) {
  int c = blockIdx.y;
  int bi = blockIdx.z;
  int p = blockIdx.x * blockDim.x + threadIdx.x;
  int hh = p >> 8, ww = p & 255;
  const float* src = in + ((size_t)bi * inRows + c) * HWPIX;
  float wk[9];
#pragma unroll
  for (int i = 0; i < 9; ++i) wk[i] = w[c * 9 + i];
  float acc = 0.f;
#pragma unroll
  for (int dy = -1; dy <= 1; ++dy) {
    int y = hh + dy;
    if (y < 0 || y >= HDIM) continue;
#pragma unroll
    for (int dx = -1; dx <= 1; ++dx) {
      int x = ww + dx;
      if (x < 0 || x >= WDIM) continue;
      acc += wk[(dy + 1) * 3 + (dx + 1)] * src[y * WDIM + x];
    }
  }
  out[((size_t)bi * outRows + c) * HWPIX + p] = acc;
}

// ---------------------------------------------------------------------------
// Attention stats stage 1: per (b, head, mode, chunk) partial 8x8 Gram +
// 8 q-norms + 8 k-norms over a 2048-pixel chunk.
// mode 0 (du): q' = q1 (y path), k' = k (x path)
// mode 1 (ud): q' = q  (x path), k' = k1 (y path)
// Deterministic: fixed-tree shfl_xor wave reduction + in-order cross-wave sum.
// ---------------------------------------------------------------------------
__global__ void stats_partial_kernel(const float* __restrict__ qkv_x,
                                     const float* __restrict__ qkv_y,
                                     float* __restrict__ part) {
  int chunk = blockIdx.x;  // 0..31
  int h = blockIdx.y;      // 0..7
  int z = blockIdx.z;      // 0..3 : b = z&1, mode = z>>1
  int bi = z & 1, mode = z >> 1;
  const float* qsrc =
      (mode == 0 ? qkv_y : qkv_x) + ((size_t)bi * 192 + h * 8) * HWPIX;
  const float* ksrc =
      (mode == 0 ? qkv_x : qkv_y) + ((size_t)bi * 192 + 64 + h * 8) * HWPIX;

  float g[64], nq[8], nk[8];
#pragma unroll
  for (int i = 0; i < 64; ++i) g[i] = 0.f;
#pragma unroll
  for (int i = 0; i < 8; ++i) { nq[i] = 0.f; nk[i] = 0.f; }

  int p0 = chunk * 2048 + threadIdx.x;
  for (int it = 0; it < 8; ++it) {
    int p = p0 + it * 256;
    float qv[8], kv[8];
#pragma unroll
    for (int d = 0; d < 8; ++d) {
      qv[d] = qsrc[(size_t)d * HWPIX + p];
      kv[d] = ksrc[(size_t)d * HWPIX + p];
    }
#pragma unroll
    for (int c = 0; c < 8; ++c)
#pragma unroll
      for (int d = 0; d < 8; ++d) g[c * 8 + d] += qv[c] * kv[d];
#pragma unroll
    for (int d = 0; d < 8; ++d) {
      nq[d] += qv[d] * qv[d];
      nk[d] += kv[d] * kv[d];
    }
  }

  __shared__ float red[8];
  int wv = threadIdx.x >> 5;
  int ln = threadIdx.x & 31;
  float* outp = part + (((size_t)z * 8 + h) * 32 + chunk) * 80;
  for (int v = 0; v < 80; ++v) {
    float val = (v < 64) ? g[v] : ((v < 72) ? nq[v - 64] : nk[v - 72]);
    val = wave_reduce_sum(val);   // fixed 5-step butterfly (deterministic)
    if (ln == 0) red[wv] = val;
    __syncthreads();
    if (threadIdx.x == 0) {
      float s = 0.f;
#pragma unroll
      for (int wvi = 0; wvi < 8; ++wvi) s += red[wvi];
      outp[v] = s;
    }
    __syncthreads();
  }
}

// Stage 2: fixed-order reduction of the 32 chunk-partials per value.
// stats layout (floats): [0]=nsq_q(128) [128]=nsq_k [256]=nsq_q1 [384]=nsq_k1
//                        [512]=G_du(1024) [1536]=G_ud(1024)
__global__ void stats_reduce_kernel(const float* __restrict__ part,
                                    float* __restrict__ stats) {
  int idx = blockIdx.x * blockDim.x + threadIdx.x;
  if (idx >= 4 * 8 * 80) return;
  int v = idx % 80;
  int h = (idx / 80) % 8;
  int z = idx / (80 * 8);
  int bi = z & 1, mode = z >> 1;
  float s = 0.f;
  for (int c = 0; c < 32; ++c)
    s += part[(((size_t)z * 8 + h) * 32 + c) * 80 + v];
  float* out;
  if (v < 64) {
    out = stats + (mode == 0 ? 512 : 1536) + ((bi * 8 + h) * 64 + v);
  } else if (v < 72) {
    out = stats + (mode == 0 ? 256 : 0) + (bi * 8 + h) * 8 + (v - 64);
  } else {
    out = stats + (mode == 0 ? 128 : 384) + (bi * 8 + h) * 8 + (v - 72);
  }
  *out = s;
}

// ---------------------------------------------------------------------------
// Normalize by row L2 norms (/ max(||.||, 1e-12)), scale by temperature,
// 8-wide softmax. Writes A_du (1024) then A_ud (1024).
// ---------------------------------------------------------------------------
__global__ void attn_softmax_kernel(const float* __restrict__ stats,
                                    const float* __restrict__ temp,
                                    float* __restrict__ A) {
  int tid = threadIdx.x;  // 0..255
  int mode = tid >> 7;
  int r = tid & 127;
  int bi = r >> 6;
  int h = (r >> 3) & 7;
  int c = r & 7;
  const float* G = stats + (mode == 0 ? 512 : 1536) + ((bi * 8 + h) * 64 + c * 8);
  const float* nqA = stats + (mode == 0 ? 256 : 0) + (bi * 8 + h) * 8;
  const float* nkA = stats + (mode == 0 ? 128 : 384) + (bi * 8 + h) * 8;
  float qn = fmaxf(sqrtf(nqA[c]), 1e-12f);
  float t = temp[h];
  float vals[8];
  float mx = -1e30f;
#pragma unroll
  for (int d = 0; d < 8; ++d) {
    float kn = fmaxf(sqrtf(nkA[d]), 1e-12f);
    float vv = G[d] / (qn * kn) * t;
    vals[d] = vv;
    mx = fmaxf(mx, vv);
  }
  float se = 0.f;
#pragma unroll
  for (int d = 0; d < 8; ++d) {
    vals[d] = expf(vals[d] - mx);
    se += vals[d];
  }
  float* out = A + mode * 1024 + ((bi * 8 + h) * 64 + c * 8);
#pragma unroll
  for (int d = 0; d < 8; ++d) out[d] = vals[d] / se;
}

// ---------------------------------------------------------------------------
// Fused: cross = A_du@v + A_ud@v1 (8x8 per head, per pixel), x_new = x+cross,
// xn2 = LN(x_new). A matrices staged through LDS.
// ---------------------------------------------------------------------------
__global__ void cross_ln2_kernel(const float* __restrict__ x,
                                 const float* __restrict__ qkv_x,
                                 const float* __restrict__ qkv_y,
                                 const float* __restrict__ A,
                                 const float* __restrict__ ln2w,
                                 const float* __restrict__ ln2b,
                                 float* __restrict__ x_new,
                                 float* __restrict__ xn2) {
  __shared__ float Adu[8][64];
  __shared__ float Aud[8][64];
  int bi = blockIdx.z;
  for (int i = threadIdx.x; i < 512; i += blockDim.x) {
    Adu[i >> 6][i & 63] = A[bi * 512 + i];
    Aud[i >> 6][i & 63] = A[1024 + bi * 512 + i];
  }
  __syncthreads();

  int p = blockIdx.x * blockDim.x + threadIdx.x;
  const float* vx = qkv_x + ((size_t)bi * 192 + 128) * HWPIX + p;
  const float* vy = qkv_y + ((size_t)bi * 192 + 128) * HWPIX + p;
  const float* xb = x + (size_t)bi * CN * HWPIX + p;

  float vals[CN];
  float s = 0.f, s2 = 0.f;
  for (int h = 0; h < 8; ++h) {
    float v0[8], v1[8];
#pragma unroll
    for (int d = 0; d < 8; ++d) {
      v0[d] = vx[(size_t)(h * 8 + d) * HWPIX];
      v1[d] = vy[(size_t)(h * 8 + d) * HWPIX];
    }
#pragma unroll
    for (int c = 0; c < 8; ++c) {
      float acc = 0.f;
#pragma unroll
      for (int d = 0; d < 8; ++d)
        acc += Adu[h][c * 8 + d] * v0[d] + Aud[h][c * 8 + d] * v1[d];
      int ch = h * 8 + c;
      float vv = xb[(size_t)ch * HWPIX] + acc;
      vals[ch] = vv;
      s += vv;
      s2 += vv * vv;
    }
  }
  float mu = s * (1.f / CN);
  float var = s2 * (1.f / CN) - mu * mu;
  float inv = rsqrtf(var + 1e-5f);
  float* xo = x_new + (size_t)bi * CN * HWPIX + p;
  float* no = xn2 + (size_t)bi * CN * HWPIX + p;
#pragma unroll
  for (int ch = 0; ch < CN; ++ch) {
    xo[(size_t)ch * HWPIX] = vals[ch];
    no[(size_t)ch * HWPIX] = (vals[ch] - mu) * inv * ln2w[ch] + ln2b[ch];
  }
}

// ---------------------------------------------------------------------------
// GDFN gate: g[c] = gelu_exact(x1[c]) * x2[c] for c<170; zero-fill K-padding
// rows 170..191 so the final GEMM runs with KPAD=192, guard-free.
// ---------------------------------------------------------------------------
__global__ void gdfn_gate_kernel(const float* __restrict__ t3,
                                 float* __restrict__ g) {
  int c = blockIdx.y;  // 0..191
  int bi = blockIdx.z;
  int p = blockIdx.x * blockDim.x + threadIdx.x;
  float* dst = g + ((size_t)bi * 192 + c) * HWPIX + p;
  if (c >= HIDN) {
    *dst = 0.f;
    return;
  }
  float x1 = t3[((size_t)bi * 340 + c) * HWPIX + p];
  float x2 = t3[((size_t)bi * 340 + HIDN + c) * HWPIX + p];
  float ge = 0.5f * x1 * (1.f + erff(x1 * 0.70710678118654752f));
  *dst = ge * x2;
}

// ---------------------------------------------------------------------------
// Workspace layout (floats)
// ---------------------------------------------------------------------------
static const size_t OFF_XN   = 0;          // 2*64*HW   =  8388608
static const size_t OFF_YN   = 8388608;
static const size_t OFF_XNEW = 16777216;
static const size_t OFF_XN2  = 25165824;
static const size_t OFF_QKVX = 33554432;   // 2*192*HW  = 25165824
static const size_t OFF_QKVY = 58720256;
static const size_t OFF_TPRE = 83886080;   // staging / g buffer: 25165824
static const size_t OFF_T2   = 109051904;  // 2*352*HW  = 46137344
static const size_t OFF_STAT = 155189248;  // partials 81920 + finals 4608
static const size_t OFF_WPAD = 155275776;  // 352*64 + 64*192 = 34816
// t3 (2*340*HW = 44564480) aliases OFF_QKVX..OFF_QKVY (50331648 floats),
// valid because v/v1 are dead after cross_ln2_kernel.

extern "C" void kernel_launch(void* const* d_in, const int* in_sizes, int n_in,
                              void* d_out, int out_size, void* d_ws,
                              size_t ws_size, hipStream_t stream) {
  const float* x          = (const float*)d_in[0];
  const float* y          = (const float*)d_in[1];
  const float* ln11_w     = (const float*)d_in[2];
  const float* ln11_b     = (const float*)d_in[3];
  const float* ln12_w     = (const float*)d_in[4];
  const float* ln12_b     = (const float*)d_in[5];
  const float* ln2_w      = (const float*)d_in[6];
  const float* ln2_b      = (const float*)d_in[7];
  const float* qkv_w      = (const float*)d_in[8];
  const float* qkv_dw     = (const float*)d_in[9];
  const float* temperature= (const float*)d_in[10];
  const float* gdfn_in_w  = (const float*)d_in[11];
  const float* gdfn_dw    = (const float*)d_in[12];
  const float* gdfn_out_w = (const float*)d_in[13];
  float* out = (float*)d_out;

  float* ws    = (float*)d_ws;
  float* xn    = ws + OFF_XN;
  float* yn    = ws + OFF_YN;
  float* x_new = ws + OFF_XNEW;
  float* xn2   = ws + OFF_XN2;
  float* qkvx  = ws + OFF_QKVX;
  float* qkvy  = ws + OFF_QKVY;
  float* tpre  = ws + OFF_TPRE;   // 1x1-conv staging, later reused as g
  float* t2    = ws + OFF_T2;
  float* part  = ws + OFF_STAT;           // 81920 floats of chunk partials
  float* stats = part + 81920;            // 4608 floats of finals
  float* Amat  = stats + 2560;            // A_du (1024) then A_ud (1024)
  float* wpadA = ws + OFF_WPAD;           // gdfn_in_w padded to [352,64]
  float* wpadB = wpadA + 352 * 64;        // gdfn_out_w padded to [64,192]
  float* t3    = qkvx;                    // alias (v dead after cross)
  float* g     = tpre;                    // alias (staging dead after dw)

  dim3 pixGrid(HWPIX / 256, 1, BN);

  // 0) zero-pad GDFN weights (removes all guards from the WMMA GEMMs)
  pad_weight_kernel<<<dim3((352 * 64 + 255) / 256), 256, 0, stream>>>(
      gdfn_in_w, wpadA, 340, 64, 64, 352 * 64);
  pad_weight_kernel<<<dim3((64 * 192 + 255) / 256), 256, 0, stream>>>(
      gdfn_out_w, wpadB, 64, 170, 192, 64 * 192);

  // 1) channel LayerNorms
  ln_ch_kernel<<<pixGrid, 256, 0, stream>>>(x, ln11_w, ln11_b, xn);
  ln_ch_kernel<<<pixGrid, 256, 0, stream>>>(y, ln12_w, ln12_b, yn);

  // 2) QKV path, x: 1x1 WMMA GEMM then depthwise 3x3
  gemm_wmma_bf16<64, false><<<dim3(HWPIX / 64, 192 / 32, BN), 128, 0, stream>>>(
      qkv_w, xn, tpre, nullptr, 192);
  dwconv3_kernel<<<dim3(HWPIX / 256, 192, BN), 256, 0, stream>>>(
      tpre, qkv_dw, qkvx, 192, 192);

  // 3) QKV path, y
  gemm_wmma_bf16<64, false><<<dim3(HWPIX / 64, 192 / 32, BN), 128, 0, stream>>>(
      qkv_w, yn, tpre, nullptr, 192);
  dwconv3_kernel<<<dim3(HWPIX / 256, 192, BN), 256, 0, stream>>>(
      tpre, qkv_dw, qkvy, 192, 192);

  // 4) attention stats: Gram matrices + row norms (deterministic 2-stage)
  stats_partial_kernel<<<dim3(32, NHN, 4), 256, 0, stream>>>(qkvx, qkvy, part);
  stats_reduce_kernel<<<dim3(10), 256, 0, stream>>>(part, stats);
  attn_softmax_kernel<<<dim3(1), 256, 0, stream>>>(stats, temperature, Amat);

  // 5) cross attention apply + residual + LN2 (fused streaming pass)
  cross_ln2_kernel<<<pixGrid, 256, 0, stream>>>(x, qkvx, qkvy, Amat, ln2_w,
                                                ln2_b, x_new, xn2);

  // 6) GDFN: in-proj WMMA GEMM (340 padded to 352), dw3x3, gate, out-proj
  gemm_wmma_bf16<64, false><<<dim3(HWPIX / 64, 352 / 32, BN), 128, 0, stream>>>(
      wpadA, xn2, t2, nullptr, 352);
  dwconv3_kernel<<<dim3(HWPIX / 256, 340, BN), 256, 0, stream>>>(
      t2, gdfn_dw, t3, 352, 340);
  gdfn_gate_kernel<<<dim3(HWPIX / 256, 192, BN), 256, 0, stream>>>(t3, g);
  gemm_wmma_bf16<192, true><<<dim3(HWPIX / 64, 64 / 32, BN), 128, 0, stream>>>(
      wpadB, g, out, x_new, 64);
}
